// BidirectionalRNN_13623636263697
// MI455X (gfx1250) — matrix operand order repaired and checked by reading, not compiled, not run on hardware
//
#include <hip/hip_runtime.h>
#include <stdint.h>

// ---------- CDNA5 WMMA types ----------
typedef __attribute__((ext_vector_type(16))) __bf16 v16bf;
typedef __attribute__((ext_vector_type(8)))  float  v8f;
typedef __attribute__((ext_vector_type(4)))  int    v4i;

union AFrag { v16bf v; uint32_t u[8]; };
union BFrag { v16bf v; uint4 q[2]; uint32_t u[8]; };
union CFrag { v8f  v; float f[8]; };

// gfx1250 async global->LDS path (probe-guarded so compile never breaks)
#if defined(__has_builtin)
#  if __has_builtin(__builtin_amdgcn_global_load_async_to_lds_b128)
#    define HAVE_ASYNC_LDS 1
#  endif
#endif
#ifndef HAVE_ASYNC_LDS
#  define HAVE_ASYNC_LDS 0
#endif

#if HAVE_ASYNC_LDS
// signature (from hipcc diagnostic): param0 = v4i addrspace(1)* (global src)
#define ASYNC_G2L_B128(gsrc, ldst)                                         \
  __builtin_amdgcn_global_load_async_to_lds_b128(                          \
      (__attribute__((address_space(1))) v4i*)(gsrc),                      \
      (__attribute__((address_space(3))) v4i*)(ldst), 0, 0)
__device__ __forceinline__ void wait_async0() {
#if defined(__has_builtin) && __has_builtin(__builtin_amdgcn_s_wait_asynccnt)
  __builtin_amdgcn_s_wait_asynccnt(0);
#else
  asm volatile("s_wait_asynccnt 0" ::: "memory");
#endif
}
#endif

// Problem dims
#define BB 32
#define SS 512
#define II 256
#define HH 512

__device__ __forceinline__ uint16_t f32_to_bf16(float f) {
  uint32_t x = __float_as_uint(f);
  uint32_t r = x + 0x7FFFu + ((x >> 16) & 1u);      // RNE
  if ((x & 0x7F800000u) == 0x7F800000u) r = x;      // inf/nan passthrough
  return (uint16_t)(r >> 16);
}

// =====================================================================
// Kernel 1: convert x and weights to bf16, fold the two bias vectors
// =====================================================================
__global__ void prep_kernel(const float* __restrict__ x,
                            const float* __restrict__ wih_f, const float* __restrict__ whh_f,
                            const float* __restrict__ bih_f, const float* __restrict__ bhh_f,
                            const float* __restrict__ wih_b, const float* __restrict__ whh_b,
                            const float* __restrict__ bih_b, const float* __restrict__ bhh_b,
                            uint16_t* __restrict__ xbf,
                            uint16_t* __restrict__ wihbf,
                            uint16_t* __restrict__ whhbf,
                            float* __restrict__ bsum)
{
  const int NX = BB * SS * II;   // 4194304
  const int NW = HH * II;        // 131072 per dir
  const int NH = HH * HH;        // 262144 per dir
  const int total = NX + 2 * NW + 2 * NH + 2 * HH;
  for (int i = blockIdx.x * blockDim.x + threadIdx.x; i < total;
       i += gridDim.x * blockDim.x) {
    if (i < NX) { xbf[i] = f32_to_bf16(x[i]); continue; }
    int j = i - NX;
    if (j < NW)     { wihbf[j] = f32_to_bf16(wih_f[j]); continue; }
    j -= NW;
    if (j < NW)     { wihbf[NW + j] = f32_to_bf16(wih_b[j]); continue; }
    j -= NW;
    if (j < NH)     { whhbf[j] = f32_to_bf16(whh_f[j]); continue; }
    j -= NH;
    if (j < NH)     { whhbf[NH + j] = f32_to_bf16(whh_b[j]); continue; }
    j -= NH;
    if (j < HH)     { bsum[j] = bih_f[j] + bhh_f[j]; continue; }
    j -= HH;
    bsum[HH + j] = bih_b[j] + bhh_b[j];
  }
}

// =====================================================================
// Kernel 2: input projection  xp[d][s][b][h] = x[b,s,:]·Wih_d[h,:] + bias
// M = B*S = 16384 (row m = b*512+s), K = I = 256, N = H = 512
// WG = 256 thr (8 waves). Tile 128(M) x 64(N), K-chunks of 32.
// =====================================================================
__launch_bounds__(256)
__global__ void gemm_xproj(const uint16_t* __restrict__ xbf,
                           const uint16_t* __restrict__ wihbf,
                           const float* __restrict__ bsum,
                           float* __restrict__ xp)
{
  __shared__ uint16_t At[128 * 32];   // 8 KB
  __shared__ uint16_t Bt[64 * 32];    // 4 KB

  const int d     = blockIdx.z;
  const int mBase = blockIdx.x * 128;
  const int nBase = blockIdx.y * 64;
  const uint16_t* W = wihbf + d * (HH * II);
  float* xpd = xp + (size_t)d * (SS * BB * HH);

  const int tid  = threadIdx.x;
  const int wave = tid >> 5;
  const int lane = tid & 31;

  CFrag acc[4];
#pragma unroll
  for (int nt = 0; nt < 4; nt++)
#pragma unroll
    for (int r = 0; r < 8; r++) acc[nt].f[r] = 0.f;

  for (int kk = 0; kk < II; kk += 32) {
    // ---- stage A tile: 128 rows x 32 bf16 (512 x 16B chunks) ----
#pragma unroll
    for (int it = 0; it < 2; it++) {
      int c = tid + it * 256;
      int row = c >> 2, ch = c & 3;
      const uint4* src = (const uint4*)(xbf + (mBase + row) * II + kk + ch * 8);
      uint4* dst = (uint4*)(At + row * 32 + ch * 8);
#if HAVE_ASYNC_LDS
      ASYNC_G2L_B128(src, dst);
#else
      *dst = *src;
#endif
    }
    // ---- stage B tile: 64 rows x 32 bf16 (256 x 16B chunks) ----
    {
      int c = tid;
      int row = c >> 2, ch = c & 3;
      const uint4* src = (const uint4*)(W + (nBase + row) * II + kk + ch * 8);
      uint4* dst = (uint4*)(Bt + row * 32 + ch * 8);
#if HAVE_ASYNC_LDS
      ASYNC_G2L_B128(src, dst);
#else
      *dst = *src;
#endif
    }
#if HAVE_ASYNC_LDS
    wait_async0();
#endif
    __syncthreads();

    // ---- A fragment: wave owns M-rows [wave*16, wave*16+16) ----
    AFrag a;
    const int rowA = wave * 16 + (lane & 15);
    const int kb8  = (lane >> 4) * 8;
#pragma unroll
    for (int v = 0; v < 8; v++) {
      int kl = ((v >= 4) ? 16 : 0) + kb8 + 2 * (v & 3);
      a.u[v] = *(const uint32_t*)(At + rowA * 32 + kl);
    }
    // ---- 4 N-subtiles ----
#pragma unroll
    for (int nt = 0; nt < 4; nt++) {
      BFrag b;
      const int col = nt * 16 + (lane & 15);
      const int ko  = (lane >> 4) * 8;
      b.q[0] = *(const uint4*)(Bt + col * 32 + ko);
      b.q[1] = *(const uint4*)(Bt + col * 32 + ko + 16);
      acc[nt].v = __builtin_amdgcn_wmma_f32_16x16x32_bf16(
          false, a.v, false, b.v, (short)0, acc[nt].v, false, false);
    }
    __syncthreads();
  }

  // ---- epilogue: add bias, store xp as (S, B, H) ----
#pragma unroll
  for (int nt = 0; nt < 4; nt++) {
#pragma unroll
    for (int r = 0; r < 8; r++) {
      int m = mBase + wave * 16 + r + ((lane >> 4) * 8);
      int n = nBase + nt * 16 + (lane & 15);
      int b = m >> 9, s = m & 511;
      xpd[(size_t)s * (BB * HH) + b * HH + n] = acc[nt].f[r] + bsum[d * HH + n];
    }
  }
}

// =====================================================================
// Kernel 3: recurrent scan. One WG (32 waves) per direction.
// Wave w owns 16 output columns; W_hh B-fragments register-resident
// (16 K-tiles x 8 VGPRs); h double-buffered in LDS (2 x 32KB).
// h_new = relu(xp[t] + h @ Whh^T), 32 WMMA + 1 barrier per step.
// =====================================================================
__launch_bounds__(1024)
__global__ void rnn_scan(const float* __restrict__ h0,
                         const uint16_t* __restrict__ whhbf,
                         const float* __restrict__ xp,
                         float* __restrict__ out)
{
  extern __shared__ uint16_t hbuf[];  // 2 * 32*512 bf16 = 64 KB

  const int d = blockIdx.x;
  const uint16_t* W = whhbf + d * (HH * HH);
  const float* xpd = xp + (size_t)d * (SS * BB * HH);
  float* hnOut = out + (size_t)BB * SS * (2 * HH) + d * (BB * HH);

  const int tid  = threadIdx.x;
  const int wave = tid >> 5;            // 0..31  -> N-subtile
  const int lane = tid & 31;
  const int nCol = wave * 16 + (lane & 15);   // this lane's output column
  const int hiM  = (lane >> 4) * 8;

  // ---- load W_hh fragments for this wave's column tile (stays in VGPRs) ----
  BFrag bfr[16];
#pragma unroll
  for (int kt = 0; kt < 16; kt++) {
    int ko = kt * 32 + ((lane >> 4) * 8);
    bfr[kt].q[0] = *(const uint4*)(W + nCol * HH + ko);
    bfr[kt].q[1] = *(const uint4*)(W + nCol * HH + ko + 16);
  }

  // ---- init h from h0[d] ----
  for (int i = tid; i < BB * HH; i += 1024)
    hbuf[i] = f32_to_bf16(h0[d * BB * HH + i]);
  __syncthreads();

  for (int t = 0; t < SS; t++) {
    const int s   = d ? (SS - 1 - t) : t;
    const int cur = (t & 1) * (BB * HH);
    const int nxt = ((t + 1) & 1) * (BB * HH);

    if (t + 1 < SS) {  // prefetch next step's xp (global_prefetch_b8)
      int s2 = d ? (SS - 2 - t) : (t + 1);
      __builtin_prefetch(xpd + (size_t)s2 * (BB * HH) + (lane & 15) * HH + nCol, 0, 1);
    }

#pragma unroll
    for (int mt = 0; mt < 2; mt++) {
      // accumulator starts at xp[t] (bias already folded in)
      CFrag c;
#pragma unroll
      for (int r = 0; r < 8; r++) {
        int m = mt * 16 + r + hiM;
        c.f[r] = xpd[(size_t)s * (BB * HH) + m * HH + nCol];
      }
      const int rowA = mt * 16 + (lane & 15);
#pragma unroll
      for (int kt = 0; kt < 16; kt++) {
        AFrag a;
        const int kb = kt * 32 + ((lane >> 4) * 8);
#pragma unroll
        for (int v = 0; v < 8; v++) {
          int kl = ((v >= 4) ? 16 : 0) + 2 * (v & 3);
          a.u[v] = *(const uint32_t*)(hbuf + cur + rowA * HH + kb + kl);
        }
        c.v = __builtin_amdgcn_wmma_f32_16x16x32_bf16(
            false, a.v, false, bfr[kt].v, (short)0, c.v, false, false);
      }
      // relu, write outputs (B,S,2H), update LDS h, final hn
#pragma unroll
      for (int r = 0; r < 8; r++) {
        float hv = c.f[r] > 0.f ? c.f[r] : 0.f;
        int m = mt * 16 + r + hiM;
        out[(size_t)m * (SS * 2 * HH) + s * (2 * HH) + d * HH + nCol] = hv;
        hbuf[nxt + m * HH + nCol] = f32_to_bf16(hv);
        if (t == SS - 1) hnOut[m * HH + nCol] = hv;
      }
    }
    __syncthreads();
  }
}

// =====================================================================
extern "C" void kernel_launch(void* const* d_in, const int* in_sizes, int n_in,
                              void* d_out, int out_size, void* d_ws, size_t ws_size,
                              hipStream_t stream)
{
  const float* x     = (const float*)d_in[0];
  const float* h0    = (const float*)d_in[1];
  const float* wih_f = (const float*)d_in[2];
  const float* whh_f = (const float*)d_in[3];
  const float* bih_f = (const float*)d_in[4];
  const float* bhh_f = (const float*)d_in[5];
  const float* wih_b = (const float*)d_in[6];
  const float* whh_b = (const float*)d_in[7];
  const float* bih_b = (const float*)d_in[8];
  const float* bhh_b = (const float*)d_in[9];
  float* out = (float*)d_out;

  // workspace layout (bytes, 16B-aligned)
  char* ws = (char*)d_ws;
  uint16_t* xbf   = (uint16_t*)(ws + 0);          //  8 MB  x in bf16
  uint16_t* wihbf = (uint16_t*)(ws + 8388608);    //  0.5MB Wih f+b bf16
  uint16_t* whhbf = (uint16_t*)(ws + 8912896);    //  1 MB  Whh f+b bf16
  float*    bsum  = (float*)   (ws + 9961472);    //  4 KB  folded biases
  float*    xp    = (float*)   (ws + 9965568);    // 64 MB  xp f32 (S,B,H) x 2

  prep_kernel<<<2048, 256, 0, stream>>>(x, wih_f, whh_f, bih_f, bhh_f,
                                        wih_b, whh_b, bih_b, bhh_b,
                                        xbf, wihbf, whhbf, bsum);
  gemm_xproj<<<dim3(128, 8, 2), 256, 0, stream>>>(xbf, wihbf, bsum, xp);
  rnn_scan<<<dim3(2), dim3(1024), 65536, stream>>>(h0, whhbf, xp, out);
}
